// interaction_module_14903536517420
// MI455X (gfx1250) — matrix operand order repaired and checked by reading, not compiled, not run on hardware
//
#include <hip/hip_runtime.h>
#include <hip/hip_bf16.h>

// ---------------------------------------------------------------------------
// out[b,o] = sum_k x[b,k]*Wa[o,k] + sum_{i<j} x[b,i]*x[b,j]*Wb[o,p(i,j)]
// Implemented as one bf16 WMMA GEMM over a unified K axis of 131328
// (130816 triangular pairs + 512 linear terms), with A generated on the fly.
// ---------------------------------------------------------------------------

typedef __bf16 bf16_t;
typedef bf16_t v16bf __attribute__((ext_vector_type(16)));
typedef float  v8f   __attribute__((ext_vector_type(8)));

#define INDIM        512
#define OUTDIM       64
#define BATCH        2048
#define NPAIR        130816      // 512*511/2
#define KTOT         131328      // NPAIR + INDIM, = 32*4104
#define NCHUNKS      4104        // KTOT / 32
#define NG8          16416       // KTOT / 8   (8-element K groups)
#define NG8_PAIR     16352       // NPAIR / 8
#define BM           64          // batch rows per workgroup
#define SPLITS       32          // K-split factor
#define CPS          129         // ceil(NCHUNKS / SPLITS)
#define GEMM_THREADS 128         // 4 waves of 32
#define XTILE_BYTES  (BM * INDIM * 2)   // 65536 B of LDS (bf16 x-tile)

// ws layout (bytes)
#define WS_BP_OFF    0u
#define WS_BP_BYTES  (NCHUNKS * 4 * 32 * 16 * 2)        // 16,809,984
#define WS_META_OFF  (WS_BP_OFF + WS_BP_BYTES)
#define WS_META_BYTES (NG8 * 4)                          // 65,664
#define WS_PART_OFF  ((WS_META_OFF + WS_META_BYTES + 255u) & ~255u)
// partials: SPLITS * BATCH * OUTDIM f32 = 16,777,216 B; total ws ~33.7 MB

// ---------------- helpers ----------------

__device__ __forceinline__ unsigned short f2bf_rne(float f) {
  unsigned int u = __float_as_uint(f);
  u += 0x7FFFu + ((u >> 16) & 1u);   // round-to-nearest-even
  return (unsigned short)(u >> 16);
}

// Bank-swizzled x-tile addressing: row m (0..63), col j (0..511).
// XOR the 8-element group index with (m & 15): 8 elems = 16 B = 4 LDS banks,
// so 16 rows at the same j land on 16 distinct 4-bank groups -> conflict free.
__device__ __forceinline__ unsigned int xswz(unsigned int m, unsigned int j) {
  return m * INDIM + ((j & 7u) | ((((j >> 3) ^ (m & 15u)) & 63u) << 3));
}

union ABreg { v16bf v; unsigned int d[8]; };
union BF2u  { unsigned int u; __hip_bfloat162 h2; };
union BF1u  { unsigned short u; __hip_bfloat16 h; };

__device__ __forceinline__ unsigned int pkmul(unsigned int a, unsigned int b) {
  BF2u x, y, r; x.u = a; y.u = b;
  r.h2 = __hmul2(x.h2, y.h2);        // packed bf16 multiply
  return r.u;
}
__device__ __forceinline__ unsigned short sclmul(unsigned short a, unsigned short b) {
  BF1u x, y, r; x.u = a; y.u = b;
  r.h = __hmul(x.h, y.h);
  return r.u;
}

// ---------------- prepass 1: pack B into WMMA lane layout (bf16) -----------
// dst element (c,t,lane,q) = Bu[32c + 16*(q/8) + 8*(lane/16) + q%8][16t + lane%16]
// flat dst index = c*2048 + t*512 + lane*16 + q
__global__ void pack_b_kernel(const float* __restrict__ Wa,
                              const float* __restrict__ Wb,
                              unsigned short* __restrict__ Bp) {
  unsigned int tid = blockIdx.x * 256u + threadIdx.x;
  if (tid >= (unsigned int)NCHUNKS * 2048u) return;
  unsigned int q    = tid & 15u;
  unsigned int lane = (tid >> 4) & 31u;
  unsigned int t    = (tid >> 9) & 3u;
  unsigned int c    = tid >> 11;
  unsigned int k = c * 32u + (q & 8u) * 2u + (lane >> 4) * 8u + (q & 7u);
  unsigned int o = t * 16u + (lane & 15u);
  float v = (k < (unsigned int)NPAIR)
              ? Wb[(size_t)o * NPAIR + k]
              : Wa[(size_t)o * INDIM + (k - NPAIR)];
  Bp[tid] = f2bf_rne(v);
}

// ---------------- prepass 2: per-8-group (i0, j0) metadata ------------------
// meta = (i0<<16)|j0 for pair groups; 0xFFFF0000|off for linear groups.
__global__ void build_meta_kernel(unsigned int* __restrict__ meta) {
  int g = blockIdx.x * 256 + threadIdx.x;
  if (g >= NG8) return;
  if (g >= NG8_PAIR) {
    meta[g] = 0xFFFF0000u | (unsigned int)((g - NG8_PAIR) * 8);
    return;
  }
  long long p = (long long)g * 8;
  // off(i) = i*(1023-i)/2 ; find largest i with off(i) <= p
  double disc = 1023.0 * 1023.0 - 8.0 * (double)p;
  int i = (int)((1023.0 - sqrt(disc)) * 0.5);
  if (i < 0) i = 0;
  if (i > 510) i = 510;
  while (i > 0 && (long long)i * (1023 - i) / 2 > p) --i;
  while (i < 510 && (long long)(i + 1) * (1023 - (i + 1)) / 2 <= p) ++i;
  long long off = (long long)i * (1023 - i) / 2;
  int j = (int)(i + 1 + (p - off));
  meta[g] = ((unsigned int)i << 16) | (unsigned int)j;
}

// ---------------- A-run generation (8 bf16 products, in registers) ----------
__device__ __forceinline__ void gen_run(unsigned int* d, const unsigned short* xl,
                                        unsigned int m, unsigned int mv) {
  unsigned int i0 = mv >> 16;
  unsigned int j0 = mv & 0xFFFFu;
  bool lin = (i0 == 0xFFFFu);
  if (lin || (j0 + 8u <= 512u)) {
    // fast path: j0..j0+7 contiguous in row i0 (or linear region, xi = 1.0)
    unsigned int xib = lin ? 0x3F80u : (unsigned int)xl[xswz(m, i0)];
    unsigned int xi2 = xib | (xib << 16);
    unsigned int g0 = j0 >> 3;
    const uint4* p0 = (const uint4*)&xl[m * INDIM + (((g0        ^ (m & 15u)) & 63u) << 3)];
    const uint4* p1 = (const uint4*)&xl[m * INDIM + ((((g0 + 1u) ^ (m & 15u)) & 63u) << 3)];
    uint4 w0 = *p0, w1 = *p1;                 // 16-elem aligned window
    unsigned int t = (j0 >> 1) & 3u;
    unsigned int aa = t==0 ? w0.x : t==1 ? w0.y : t==2 ? w0.z : w0.w;
    unsigned int bb = t==0 ? w0.y : t==1 ? w0.z : t==2 ? w0.w : w1.x;
    unsigned int cc = t==0 ? w0.z : t==1 ? w0.w : t==2 ? w1.x : w1.y;
    unsigned int dd = t==0 ? w0.w : t==1 ? w1.x : t==2 ? w1.y : w1.z;
    unsigned int ee = t==0 ? w1.x : t==1 ? w1.y : t==2 ? w1.z : w1.w;
    if (j0 & 1u) {                            // 16-bit funnel shift
      aa = (aa >> 16) | (bb << 16);
      bb = (bb >> 16) | (cc << 16);
      cc = (cc >> 16) | (dd << 16);
      dd = (dd >> 16) | (ee << 16);
    }
    d[0] = pkmul(aa, xi2);
    d[1] = pkmul(bb, xi2);
    d[2] = pkmul(cc, xi2);
    d[3] = pkmul(dd, xi2);
  } else {
    // slow path (~6% of groups): walk across triangle-row boundaries
    unsigned int i = i0, j = j0;
#pragma unroll
    for (int q = 0; q < 4; ++q) {
      unsigned int lo = (unsigned int)sclmul(xl[xswz(m, i)], xl[xswz(m, j)]);
      ++j; if (j == 512u) { ++i; j = i + 1u; }
      unsigned int hi = (unsigned int)sclmul(xl[xswz(m, i)], xl[xswz(m, j)]);
      ++j; if (j == 512u) { ++i; j = i + 1u; }
      d[q] = lo | (hi << 16);
    }
  }
}

// ---------------- main split-K WMMA GEMM ------------------------------------
__global__ __launch_bounds__(GEMM_THREADS) void
gemm_kernel(const float* __restrict__ x, const unsigned short* __restrict__ Bp,
            const unsigned int* __restrict__ meta, float* __restrict__ part) {
  extern __shared__ unsigned short xl[];     // 64 x 512 bf16, swizzled
  const unsigned int tid = threadIdx.x;
  const unsigned int mb = blockIdx.x, sp = blockIdx.y;

  // cooperative load+convert of the x tile
  {
    unsigned int r  = tid >> 1;
    unsigned int c0 = (tid & 1u) * 256u;
    const float4* src = (const float4*)(x + (size_t)(mb * BM + r) * INDIM + c0);
#pragma unroll 4
    for (unsigned int k = 0; k < 64u; ++k) {
      float4 v = src[k];
      uint2 pk;
      pk.x = (unsigned int)f2bf_rne(v.x) | ((unsigned int)f2bf_rne(v.y) << 16);
      pk.y = (unsigned int)f2bf_rne(v.z) | ((unsigned int)f2bf_rne(v.w) << 16);
      *(uint2*)&xl[xswz(r, c0 + 4u * k)] = pk;
    }
  }
  __syncthreads();

  const unsigned int lane = tid & 31u;
  const unsigned int wv   = tid >> 5;
  const unsigned int half = lane >> 4;          // which 16-lane group
  const unsigned int m    = wv * 16u + (lane & 15u);   // local A row

  v8f acc0 = {0,0,0,0,0,0,0,0};
  v8f acc1 = acc0, acc2 = acc0, acc3 = acc0;

  unsigned int c0 = sp * CPS;
  unsigned int c1 = c0 + CPS; if (c1 > NCHUNKS) c1 = NCHUNKS;

  const uint4* bbase = ((const uint4*)Bp) + (size_t)lane * 2u;   // lane's 32B run

  for (unsigned int c = c0; c < c1; ++c) {
    // A operand: element e (0..7) -> K = (half?8:0)+e ; e (8..15) -> 16+(half?8:0)+(e-8)
    unsigned int mv0 = meta[c * 4u + half];
    unsigned int mv1 = meta[c * 4u + 2u + half];
    ABreg a;
    gen_run(&a.d[0], xl, m, mv0);
    gen_run(&a.d[4], xl, m, mv1);

    const uint4* bq = bbase + (size_t)c * 256u;   // 4096 B per chunk
    __builtin_prefetch((const void*)(bq + 256u), 0, 1);  // next chunk of B

    ABreg b0, b1, b2, b3;
    uint4 q;
    q = bq[0];     b0.d[0]=q.x; b0.d[1]=q.y; b0.d[2]=q.z; b0.d[3]=q.w;
    q = bq[1];     b0.d[4]=q.x; b0.d[5]=q.y; b0.d[6]=q.z; b0.d[7]=q.w;
    q = bq[64];    b1.d[0]=q.x; b1.d[1]=q.y; b1.d[2]=q.z; b1.d[3]=q.w;
    q = bq[65];    b1.d[4]=q.x; b1.d[5]=q.y; b1.d[6]=q.z; b1.d[7]=q.w;
    q = bq[128];   b2.d[0]=q.x; b2.d[1]=q.y; b2.d[2]=q.z; b2.d[3]=q.w;
    q = bq[129];   b2.d[4]=q.x; b2.d[5]=q.y; b2.d[6]=q.z; b2.d[7]=q.w;
    q = bq[192];   b3.d[0]=q.x; b3.d[1]=q.y; b3.d[2]=q.z; b3.d[3]=q.w;
    q = bq[193];   b3.d[4]=q.x; b3.d[5]=q.y; b3.d[6]=q.z; b3.d[7]=q.w;

    acc0 = __builtin_amdgcn_wmma_f32_16x16x32_bf16(false, a.v, false, b0.v,
                                                   (short)0, acc0, false, false);
    acc1 = __builtin_amdgcn_wmma_f32_16x16x32_bf16(false, a.v, false, b1.v,
                                                   (short)0, acc1, false, false);
    acc2 = __builtin_amdgcn_wmma_f32_16x16x32_bf16(false, a.v, false, b2.v,
                                                   (short)0, acc2, false, false);
    acc3 = __builtin_amdgcn_wmma_f32_16x16x32_bf16(false, a.v, false, b3.v,
                                                   (short)0, acc3, false, false);
  }

  // C layout: lane holds col = lane%16, rows v + 8*half (v = 0..7)
  const unsigned int grow = mb * BM + wv * 16u + half * 8u;
  float* pout = part + ((size_t)sp * BATCH + grow) * OUTDIM + (lane & 15u);
#pragma unroll
  for (int v = 0; v < 8; ++v) {
    pout[(size_t)v * OUTDIM +  0] = acc0[v];
    pout[(size_t)v * OUTDIM + 16] = acc1[v];
    pout[(size_t)v * OUTDIM + 32] = acc2[v];
    pout[(size_t)v * OUTDIM + 48] = acc3[v];
  }
}

// ---------------- deterministic split-K reduction ---------------------------
__global__ void reduce_kernel(const float* __restrict__ part, float* __restrict__ out) {
  unsigned int t = blockIdx.x * 256u + threadIdx.x;   // BATCH*OUTDIM threads
  float s = 0.f;
#pragma unroll
  for (int k = 0; k < SPLITS; ++k)
    s += part[(size_t)k * (BATCH * OUTDIM) + t];
  out[t] = s;
}

// ---------------- host side -------------------------------------------------
extern "C" void kernel_launch(void* const* d_in, const int* in_sizes, int n_in,
                              void* d_out, int out_size, void* d_ws, size_t ws_size,
                              hipStream_t stream) {
  (void)in_sizes; (void)n_in; (void)out_size; (void)ws_size;
  const float* barray = (const float*)d_in[0];   // [2048, 512]
  const float* Wa     = (const float*)d_in[1];   // [64, 512]
  const float* Wb     = (const float*)d_in[2];   // [64, 130816]
  float* out = (float*)d_out;                    // [2048, 64]

  char* ws = (char*)d_ws;                        // needs ~33.7 MB
  unsigned short* Bp   = (unsigned short*)(ws + WS_BP_OFF);
  unsigned int*   meta = (unsigned int*)(ws + WS_META_OFF);
  float*          part = (float*)(ws + WS_PART_OFF);

  pack_b_kernel<<<dim3((NCHUNKS * 2048u) / 256u), 256, 0, stream>>>(Wa, Wb, Bp);
  build_meta_kernel<<<dim3((NG8 + 255) / 256), 256, 0, stream>>>(meta);
  gemm_kernel<<<dim3(BATCH / BM, SPLITS), GEMM_THREADS, XTILE_BYTES, stream>>>(
      barray, Bp, meta, part);
  reduce_kernel<<<dim3((BATCH * OUTDIM) / 256), 256, 0, stream>>>(part, out);
}
